// LinearAttentionLayer_75780402970796
// MI455X (gfx1250) — matrix operand order repaired and checked by reading, not compiled, not run on hardware
//
#include <hip/hip_runtime.h>
#include <hip/hip_bf16.h>

#define DM   1024
#define LSEQ 2048
#define NB   4
#define ROWS (NB*LSEQ)   // 8192
#define CH   64          // attention chunk length
#define NC   (LSEQ/CH)   // 32 chunks
#define NH   (NB*16)     // 64 (batch*heads)

typedef __attribute__((ext_vector_type(16))) _Float16 v16h;
typedef __attribute__((ext_vector_type(8)))  _Float16 h8;
typedef __attribute__((ext_vector_type(4)))  _Float16 h4;
typedef __attribute__((ext_vector_type(8)))  float    v8f;
typedef __attribute__((ext_vector_type(4)))  int      i4;

// ---- async global->LDS copy (ASYNCcnt path), with synchronous fallback ----
#if defined(__has_builtin)
#if __has_builtin(__builtin_amdgcn_global_load_async_to_lds_b128) && \
    __has_builtin(__builtin_amdgcn_s_wait_asynccnt)
#define HAVE_ASYNC_LDS 1
#endif
#endif
#ifndef HAVE_ASYNC_LDS
#define HAVE_ASYNC_LDS 0
#endif

__device__ __forceinline__ void async_copy_b128(void* lds_dst, const void* gsrc) {
#if HAVE_ASYNC_LDS
  // generic LDS pointer's low 32 bits are the LDS byte offset (aperture rule)
  auto l = (__attribute__((address_space(3))) i4*)(unsigned long long)
               (unsigned)(unsigned long long)(uintptr_t)lds_dst;
  auto g = (__attribute__((address_space(1))) i4*)(uintptr_t)gsrc;
  __builtin_amdgcn_global_load_async_to_lds_b128(g, l, 0, 0);
#else
  *(h8*)lds_dst = *(const h8*)gsrc;
#endif
}

__device__ __forceinline__ void async_join() {
#if HAVE_ASYNC_LDS
  __builtin_amdgcn_s_wait_asynccnt(0);
#endif
}

__device__ __forceinline__ v8f wmma32(v16h a, v16h b, v8f c) {
  return __builtin_amdgcn_wmma_f32_16x16x32_f16(false, a, false, b, (short)0, c,
                                                false, false);
}

// A fragment, 16x32 f16, from row-major LDS A[m][k] (ld halves per row).
// Per ISA 7.12.2: lane(m=lane&15,h=lane>>4): VGPR0-3 hold K=8h+0..7,
// VGPR4-7 hold K=16+8h+0..7 -> two contiguous 8-half (16B) runs.
__device__ __forceinline__ v16h frag_a_rm(const _Float16* A, int mbase, int k0, int ld) {
  const int lane = threadIdx.x & 31;
  const int mm = lane & 15, hh = lane >> 4;
  const _Float16* p = A + (size_t)(mbase + mm) * ld + k0;
  h8 lo = *(const h8*)(p + 8 * hh);
  h8 hi = *(const h8*)(p + 16 + 8 * hh);
  v16h r;
#pragma unroll
  for (int j = 0; j < 8; ++j) { r[j] = lo[j]; r[j + 8] = hi[j]; }
  return r;
}

// B fragment, 32x16 f16, from transposed LDS image BT[n][k] (ld halves per row).
// lane(n=lane&15,h=lane>>4) needs rows K = 8h..8h+7 and 16+8h..16+8h+7 of column n:
// two 8-half runs of BT row n (same K order as frag_a_rm).
__device__ __forceinline__ v16h frag_b_ct(const _Float16* BT, int nbase, int k0, int ld) {
  const int lane = threadIdx.x & 31;
  const int nn = lane & 15, hh = lane >> 4;
  const _Float16* p = BT + (size_t)(nbase + nn) * ld + k0;
  h8 lo = *(const h8*)(p + 8 * hh);
  h8 hi = *(const h8*)(p + 16 + 8 * hh);
  v16h r;
#pragma unroll
  for (int j = 0; j < 8; ++j) { r[j] = lo[j]; r[j + 8] = hi[j]; }
  return r;
}

__device__ __forceinline__ float elu1(float x) {
  return x > 0.f ? x + 1.f : __expf(x);
}

// ---------------------------------------------------------------------------
// Kernel 1: C = act(X[f32] @ W[f32] + b) -> f16. Block tile 128x64, 8 waves
// (4x2), each wave 32x32 (4 accumulators). K-step = 32.
// ---------------------------------------------------------------------------
__global__ void proj_gemm_kernel(const float* __restrict__ X,
                                 const float* __restrict__ W,
                                 const float* __restrict__ bias,
                                 _Float16* __restrict__ out, int act) {
  __shared__ __align__(16) _Float16 lds_a[128 * 32];   // A tile row-major
  __shared__ __align__(16) _Float16 lds_wt[64 * 32];   // W tile transposed [n][k]
  const int m0 = blockIdx.x * 128, n0 = blockIdx.y * 64;
  const int tid = threadIdx.x;
  const int w = tid >> 5, lane = tid & 31;
  const int wm = w >> 1, wn = w & 1;
  const int r8 = lane >> 4, nn = lane & 15;

  v8f acc[2][2];
#pragma unroll
  for (int mi = 0; mi < 2; ++mi)
#pragma unroll
    for (int ni = 0; ni < 2; ++ni) {
      v8f z = {0.f, 0.f, 0.f, 0.f, 0.f, 0.f, 0.f, 0.f};
      acc[mi][ni] = z;
    }

  for (int kk = 0; kk < DM / 32; ++kk) {
    // stage A: 128x32 f32 -> f16, row-major (conversion => VALU path)
#pragma unroll
    for (int it = 0; it < 4; ++it) {
      int i = tid + it * 256;
      int row = i >> 3, c4 = i & 7;
      float4 val = *(const float4*)(X + (size_t)(m0 + row) * DM + kk * 32 + c4 * 4);
      h4 hv = {(_Float16)val.x, (_Float16)val.y, (_Float16)val.z, (_Float16)val.w};
      *(h4*)(lds_a + row * 32 + c4 * 4) = hv;
    }
    // stage W^T: W[k][n] -> lds_wt[n][k]
#pragma unroll
    for (int it = 0; it < 2; ++it) {
      int i = tid + it * 256;
      int row = i >> 4, c4 = i & 15;
      float4 val = *(const float4*)(W + (size_t)(kk * 32 + row) * DM + n0 + c4 * 4);
      lds_wt[(c4 * 4 + 0) * 32 + row] = (_Float16)val.x;
      lds_wt[(c4 * 4 + 1) * 32 + row] = (_Float16)val.y;
      lds_wt[(c4 * 4 + 2) * 32 + row] = (_Float16)val.z;
      lds_wt[(c4 * 4 + 3) * 32 + row] = (_Float16)val.w;
    }
    __syncthreads();

    v16h a0 = frag_a_rm(lds_a, wm * 32, 0, 32);
    v16h a1 = frag_a_rm(lds_a, wm * 32 + 16, 0, 32);
    v16h b0 = frag_b_ct(lds_wt, wn * 32, 0, 32);
    v16h b1 = frag_b_ct(lds_wt, wn * 32 + 16, 0, 32);
    acc[0][0] = wmma32(a0, b0, acc[0][0]);
    acc[0][1] = wmma32(a0, b1, acc[0][1]);
    acc[1][0] = wmma32(a1, b0, acc[1][0]);
    acc[1][1] = wmma32(a1, b1, acc[1][1]);
    __syncthreads();
    if (kk + 1 < DM / 32)
      __builtin_prefetch(X + (size_t)(m0 + (tid >> 1)) * DM + (kk + 1) * 32, 0, 1);
  }

#pragma unroll
  for (int mi = 0; mi < 2; ++mi)
#pragma unroll
    for (int ni = 0; ni < 2; ++ni)
#pragma unroll
      for (int r = 0; r < 8; ++r) {
        int m = m0 + wm * 32 + mi * 16 + r + 8 * r8;
        int ncol = n0 + wn * 32 + ni * 16 + nn;
        float v = acc[mi][ni][r] + bias[ncol];
        if (act) v = elu1(v);
        out[(size_t)m * DM + ncol] = (_Float16)v;
      }
}

// ---------------------------------------------------------------------------
// Kernel 2: per (n,h,chunk): S_c[d][k] = sum_t kp[t][d]*v[t][k] (64x64 WMMA),
// Zsum_c[d] = sum_t qp[t][d]. Block = 128 threads (4 waves, one d-tile each).
// ---------------------------------------------------------------------------
__global__ void chunk_kv_kernel(const _Float16* __restrict__ qp,
                                const _Float16* __restrict__ kp,
                                const _Float16* __restrict__ vp,
                                float* __restrict__ S, float* __restrict__ Z) {
  __shared__ __align__(16) _Float16 lds_kT[64 * CH];  // [d][t]
  __shared__ __align__(16) _Float16 lds_vT[64 * CH];  // [k][t]
  const int c = blockIdx.x, nh = blockIdx.y;
  const int n = nh >> 4, hh = nh & 15;
  const size_t rowbase = (size_t)n * LSEQ + (size_t)c * CH;
  const int tid = threadIdx.x;

#pragma unroll
  for (int it = 0; it < 32; ++it) {
    int i = tid + it * 128;
    int t = i >> 6, d = i & 63;
    size_t g = (rowbase + t) * DM + hh * 64 + d;
    lds_kT[d * CH + t] = kp[g];
    lds_vT[d * CH + t] = vp[g];
  }
  __syncthreads();

  const int w = tid >> 5, lane = tid & 31;
  const int r8 = lane >> 4, nn = lane & 15;
  v8f acc[4];
#pragma unroll
  for (int j = 0; j < 4; ++j) {
    v8f z = {0.f, 0.f, 0.f, 0.f, 0.f, 0.f, 0.f, 0.f};
    acc[j] = z;
  }
#pragma unroll
  for (int t0 = 0; t0 < CH; t0 += 32) {
    v16h a = frag_a_rm(lds_kT, w * 16, t0, CH);
#pragma unroll
    for (int j = 0; j < 4; ++j) {
      v16h b = frag_b_ct(lds_vT, j * 16, t0, CH);
      acc[j] = wmma32(a, b, acc[j]);
    }
  }
  const size_t base = ((size_t)nh * NC + c) * 4096;
#pragma unroll
  for (int j = 0; j < 4; ++j)
#pragma unroll
    for (int r = 0; r < 8; ++r) {
      int d = w * 16 + r + 8 * r8;
      int k = j * 16 + nn;
      S[base + d * 64 + k] = acc[j][r];
    }
  if (tid < 64) {
    float s = 0.f;
    for (int t = 0; t < CH; ++t)
      s += (float)qp[(rowbase + t) * DM + hh * 64 + tid];
    Z[((size_t)nh * NC + c) * 64 + tid] = s;
  }
}

// ---------------------------------------------------------------------------
// Kernel 3: in-place exclusive prefix over chunks for S (64x64) and Z (64),
// per (n,h). 64 blocks x 256 threads; serial over 32 chunks only.
// ---------------------------------------------------------------------------
__global__ void prefix_kernel(float* __restrict__ S, float* __restrict__ Z) {
  const int nh = blockIdx.x;
  float run[16];
#pragma unroll
  for (int e = 0; e < 16; ++e) run[e] = 0.f;
  for (int c = 0; c < NC; ++c) {
    size_t base = ((size_t)nh * NC + c) * 4096;
#pragma unroll
    for (int e = 0; e < 16; ++e) {
      size_t idx = base + (size_t)e * 256 + threadIdx.x;
      float t = S[idx];
      S[idx] = run[e];
      run[e] += t;
    }
  }
  if (threadIdx.x < 64) {
    float r = 0.f;
    for (int c = 0; c < NC; ++c) {
      size_t idx = ((size_t)nh * NC + c) * 64 + threadIdx.x;
      float t = Z[idx];
      Z[idx] = r;
      r += t;
    }
  }
}

// ---------------------------------------------------------------------------
// Kernel 4: per (n,h,chunk): O = (Q@S_pref + mask(Q@K^T)@V) / Zcum -> f16.
// Block = 128 threads (4 waves, one 16-row t-strip each).
// q/k tiles are straight row-major copies -> async global->LDS path.
// ---------------------------------------------------------------------------
__global__ void attn_kernel(const _Float16* __restrict__ qp,
                            const _Float16* __restrict__ kp,
                            const _Float16* __restrict__ vp,
                            const float* __restrict__ S,
                            const float* __restrict__ Z,
                            _Float16* __restrict__ attn) {
  __shared__ __align__(16) _Float16 lds_q[CH * 64];   // [t][d]
  __shared__ __align__(16) _Float16 lds_k[CH * 64];   // [s][d] (== BT for Q@K^T)
  __shared__ __align__(16) _Float16 lds_vT[64 * CH];  // [k][s]
  __shared__ __align__(16) _Float16 lds_ST[64 * 64];  // [k][d] (S transposed)
  __shared__ __align__(16) _Float16 lds_P[CH * CH];   // [t][s] masked scores
  __shared__ __align__(16) float    lds_Z[CH * 64];   // [t][d] cumulative Z
  const int c = blockIdx.x, nh = blockIdx.y;
  const int n = nh >> 4, hh = nh & 15;
  const size_t rowbase = (size_t)n * LSEQ + (size_t)c * CH;
  const int tid = threadIdx.x;
  const size_t Sbase = ((size_t)nh * NC + c) * 4096;

  // straight copies: q and k tiles, 8-half granules, async to LDS
#pragma unroll
  for (int it = 0; it < 4; ++it) {
    int i = tid + it * 128;                    // 512 granules of 8 halves
    int t = i >> 3, c8 = i & 7;
    size_t g = (rowbase + t) * DM + hh * 64 + c8 * 8;
    async_copy_b128(lds_q + t * 64 + c8 * 8, qp + g);
    async_copy_b128(lds_k + t * 64 + c8 * 8, kp + g);
  }
  // transposed / converted stages stay on the VALU path
#pragma unroll
  for (int it = 0; it < 32; ++it) {
    int i = tid + it * 128;
    int t = i >> 6, d = i & 63;
    lds_vT[d * CH + t] = vp[(rowbase + t) * DM + hh * 64 + d];
    lds_ST[d * 64 + t] = (_Float16)S[Sbase + (size_t)t * 64 + d];
  }
  async_join();
  __syncthreads();

  const int w = tid >> 5, lane = tid & 31;
  const int r8 = lane >> 4, nn = lane & 15;

  // P = mask(Q @ K^T)
  {
    v8f p[4];
#pragma unroll
    for (int j = 0; j < 4; ++j) {
      v8f z = {0.f, 0.f, 0.f, 0.f, 0.f, 0.f, 0.f, 0.f};
      p[j] = z;
    }
#pragma unroll
    for (int d0 = 0; d0 < 64; d0 += 32) {
      v16h a = frag_a_rm(lds_q, w * 16, d0, 64);
#pragma unroll
      for (int j = 0; j < 4; ++j) {
        v16h b = frag_b_ct(lds_k, j * 16, d0, 64);
        p[j] = wmma32(a, b, p[j]);
      }
    }
#pragma unroll
    for (int j = 0; j < 4; ++j)
#pragma unroll
      for (int r = 0; r < 8; ++r) {
        int t = w * 16 + r + 8 * r8;
        int s = j * 16 + nn;
        lds_P[t * CH + s] = (_Float16)((s <= t) ? p[j][r] : 0.f);
      }
  }

  // Zcum[t][d] = Zprefix[d] + inclusive cumsum_t qp[t][d]
  if (tid < 64) {
    float run = Z[((size_t)nh * NC + c) * 64 + tid];
    for (int t = 0; t < CH; ++t) {
      run += (float)lds_q[t * 64 + tid];
      lds_Z[t * 64 + tid] = run;
    }
  }
  __syncthreads();

  // O = Q @ S_pref + P @ V
  v8f o[4];
#pragma unroll
  for (int j = 0; j < 4; ++j) {
    v8f z = {0.f, 0.f, 0.f, 0.f, 0.f, 0.f, 0.f, 0.f};
    o[j] = z;
  }
#pragma unroll
  for (int d0 = 0; d0 < 64; d0 += 32) {
    v16h a = frag_a_rm(lds_q, w * 16, d0, 64);
#pragma unroll
    for (int j = 0; j < 4; ++j) {
      v16h b = frag_b_ct(lds_ST, j * 16, d0, 64);
      o[j] = wmma32(a, b, o[j]);
    }
  }
#pragma unroll
  for (int s0 = 0; s0 < CH; s0 += 32) {
    v16h a = frag_a_rm(lds_P, w * 16, s0, CH);
#pragma unroll
    for (int j = 0; j < 4; ++j) {
      v16h b = frag_b_ct(lds_vT, j * 16, s0, CH);
      o[j] = wmma32(a, b, o[j]);
    }
  }

#pragma unroll
  for (int j = 0; j < 4; ++j)
#pragma unroll
    for (int r = 0; r < 8; ++r) {
      int t = w * 16 + r + 8 * r8;
      int k = j * 16 + nn;
      float val = o[j][r] / lds_Z[t * 64 + k];
      attn[(rowbase + t) * DM + hh * 64 + k] = (_Float16)val;
    }
}

// ---------------------------------------------------------------------------
// Kernel 5: final GEMM: out[f32] = attn[f16] @ Wo + bo. Same tiling as K1;
// A tile (already f16, row-major) staged via async global->LDS.
// ---------------------------------------------------------------------------
__global__ void out_gemm_kernel(const _Float16* __restrict__ X,
                                const float* __restrict__ W,
                                const float* __restrict__ bias,
                                float* __restrict__ out) {
  __shared__ __align__(16) _Float16 lds_a[128 * 32];
  __shared__ __align__(16) _Float16 lds_wt[64 * 32];
  const int m0 = blockIdx.x * 128, n0 = blockIdx.y * 64;
  const int tid = threadIdx.x;
  const int w = tid >> 5, lane = tid & 31;
  const int wm = w >> 1, wn = w & 1;
  const int r8 = lane >> 4, nn = lane & 15;

  v8f acc[2][2];
#pragma unroll
  for (int mi = 0; mi < 2; ++mi)
#pragma unroll
    for (int ni = 0; ni < 2; ++ni) {
      v8f z = {0.f, 0.f, 0.f, 0.f, 0.f, 0.f, 0.f, 0.f};
      acc[mi][ni] = z;
    }

  for (int kk = 0; kk < DM / 32; ++kk) {
    // stage A (f16, straight copy): 512 granules of 8 halves, async
#pragma unroll
    for (int it = 0; it < 2; ++it) {
      int i = tid + it * 256;
      int row = i >> 2, c8 = i & 3;
      async_copy_b128(lds_a + row * 32 + c8 * 8,
                      X + (size_t)(m0 + row) * DM + kk * 32 + c8 * 8);
    }
    // stage W^T (conversion+transpose => VALU path)
#pragma unroll
    for (int it = 0; it < 2; ++it) {
      int i = tid + it * 256;
      int row = i >> 4, c4 = i & 15;
      float4 val = *(const float4*)(W + (size_t)(kk * 32 + row) * DM + n0 + c4 * 4);
      lds_wt[(c4 * 4 + 0) * 32 + row] = (_Float16)val.x;
      lds_wt[(c4 * 4 + 1) * 32 + row] = (_Float16)val.y;
      lds_wt[(c4 * 4 + 2) * 32 + row] = (_Float16)val.z;
      lds_wt[(c4 * 4 + 3) * 32 + row] = (_Float16)val.w;
    }
    async_join();
    __syncthreads();

    v16h a0 = frag_a_rm(lds_a, wm * 32, 0, 32);
    v16h a1 = frag_a_rm(lds_a, wm * 32 + 16, 0, 32);
    v16h b0 = frag_b_ct(lds_wt, wn * 32, 0, 32);
    v16h b1 = frag_b_ct(lds_wt, wn * 32 + 16, 0, 32);
    acc[0][0] = wmma32(a0, b0, acc[0][0]);
    acc[0][1] = wmma32(a0, b1, acc[0][1]);
    acc[1][0] = wmma32(a1, b0, acc[1][0]);
    acc[1][1] = wmma32(a1, b1, acc[1][1]);
    __syncthreads();
    if (kk + 1 < DM / 32)
      __builtin_prefetch(X + (size_t)(m0 + (tid >> 1)) * DM + (kk + 1) * 32, 0, 1);
  }

#pragma unroll
  for (int mi = 0; mi < 2; ++mi)
#pragma unroll
    for (int ni = 0; ni < 2; ++ni)
#pragma unroll
      for (int r = 0; r < 8; ++r) {
        int m = m0 + wm * 32 + mi * 16 + r + 8 * r8;
        int ncol = n0 + wn * 32 + ni * 16 + nn;
        out[(size_t)m * DM + ncol] = acc[mi][ni][r] + bias[ncol];
      }
}

// ---------------------------------------------------------------------------
extern "C" void kernel_launch(void* const* d_in, const int* in_sizes, int n_in,
                              void* d_out, int out_size, void* d_ws, size_t ws_size,
                              hipStream_t stream) {
  (void)in_sizes; (void)n_in; (void)out_size; (void)ws_size;
  const float* queries = (const float*)d_in[0];
  const float* keys    = (const float*)d_in[1];
  const float* values  = (const float*)d_in[2];
  const float* Wq = (const float*)d_in[3];  const float* bq = (const float*)d_in[4];
  const float* Wk = (const float*)d_in[5];  const float* bk = (const float*)d_in[6];
  const float* Wv = (const float*)d_in[7];  const float* bv = (const float*)d_in[8];
  const float* Wo = (const float*)d_in[9];  const float* bo = (const float*)d_in[10];
  float* out = (float*)d_out;

  // workspace layout (f16 activations + f32 chunk state): ~98 MB
  _Float16* qp   = (_Float16*)d_ws;
  _Float16* kp   = qp + (size_t)ROWS * DM;
  _Float16* vp   = kp + (size_t)ROWS * DM;
  _Float16* attn = vp + (size_t)ROWS * DM;
  float* S  = (float*)(attn + (size_t)ROWS * DM);
  float* Zc = S + (size_t)NH * NC * 4096;

  dim3 g1(ROWS / 128, DM / 64);
  proj_gemm_kernel<<<g1, 256, 0, stream>>>(queries, Wq, bq, qp, 1);
  proj_gemm_kernel<<<g1, 256, 0, stream>>>(keys,    Wk, bk, kp, 1);
  proj_gemm_kernel<<<g1, 256, 0, stream>>>(values,  Wv, bv, vp, 0);
  chunk_kv_kernel<<<dim3(NC, NH), 128, 0, stream>>>(qp, kp, vp, S, Zc);
  prefix_kernel<<<NH, 256, 0, stream>>>(S, Zc);
  attn_kernel<<<dim3(NC, NH), 128, 0, stream>>>(qp, kp, vp, S, Zc, attn);
  out_gemm_kernel<<<g1, 256, 0, stream>>>(attn, Wo, bo, out);
}
